// BEVPool_2628519985965
// MI455X (gfx1250) — matrix operand order repaired and checked by reading, not compiled, not run on hardware
//
#include <hip/hip_runtime.h>

// ---------------------------------------------------------------------------
// Types for CDNA5 WMMA (wave32): bf16 A/B fragments (8 VGPRs), f32 acc (8 VGPRs)
// ---------------------------------------------------------------------------
typedef __bf16 v16bf __attribute__((ext_vector_type(16)));
typedef float  v8f   __attribute__((ext_vector_type(8)));

union FragB16 { v16bf v; uint4 q[2]; };

__device__ __forceinline__ unsigned short f2bf(float f) {
  unsigned u = __float_as_uint(f);
  u += 0x7fffu + ((u >> 16) & 1u);            // round-to-nearest-even
  return (unsigned short)(u >> 16);
}

// ---------------------------------------------------------------------------
// Problem constants
// ---------------------------------------------------------------------------
#define NB   4
#define CCH  64
#define DD   2
#define HH   200
#define WW   176
#define PP   (HH * WW)            // 35200
#define TILE 128                  // spatial positions per workgroup
#define ROWS (TILE * DD)          // 256 GEMM rows per workgroup
#define LDSTR 72                  // padded bf16 row stride (bank-conflict free b128)

// ws byte offsets
#define OFF_F1   ((size_t)0)                               // f1 buffer: 4*128*35200 f32
#define F1_BYTES ((size_t)NB * 128 * PP * 4)               // 72,089,600
#define OFF_W1F  (F1_BYTES)                                // 4096 bf16 (A frags, W1)
#define OFF_W2F  (OFF_W1F + 8192)                          // 2048 bf16 (A frags, W2)
#define OFF_PED  (OFF_W2F + 4096)                          // 2*64 f32 (PE-d + b1)
#define OFF_PEH  (OFF_PED + 512)                           // 200*64 f32
#define OFF_PEW  (OFF_PEH + 51200)                         // 176*64 f32
#define OFF_S1   (OFF_PEW + 45056)                         // 64 f32
#define OFF_T1   (OFF_S1 + 256)                            // 64 f32
#define OFF_S2   (OFF_T1 + 256)                            // 32 f32
#define OFF_T2   (OFF_S2 + 128)                            // 32 f32 (be2 + b2*s2)
#define OFF_W3   (OFF_T2 + 128)                            // 32 f32
#define OFF_B3   (OFF_W3 + 128)                            // 1 f32

// A-fragment K index for 16-bit WMMA A (16x32), contiguous k order used here (B reads match):
//   lane = half*16+row ; value i in 0..15 ; K = 32*ks + half*16 + i
__device__ __forceinline__ int afrag_k(int ks, int half, int i) {
  return 32 * ks + half * 16 + i;
}

// ---------------------------------------------------------------------------
// Kernel 0: precompute weight fragments, separable PE biases, folded BN consts
// ---------------------------------------------------------------------------
__global__ void prep_kernel(const float* __restrict__ w1, const float* __restrict__ b1,
                            const float* __restrict__ g1, const float* __restrict__ be1,
                            const float* __restrict__ w2, const float* __restrict__ b2,
                            const float* __restrict__ g2, const float* __restrict__ be2,
                            const float* __restrict__ w3, const float* __restrict__ b3,
                            char* __restrict__ ws) {
  unsigned short* w1f = (unsigned short*)(ws + OFF_W1F);
  unsigned short* w2f = (unsigned short*)(ws + OFF_W2F);
  float* ped = (float*)(ws + OFF_PED);
  float* peh = (float*)(ws + OFF_PEH);
  float* pew = (float*)(ws + OFF_PEW);
  float* s1  = (float*)(ws + OFF_S1);
  float* t1  = (float*)(ws + OFF_T1);
  float* s2  = (float*)(ws + OFF_S2);
  float* t2  = (float*)(ws + OFF_T2);
  float* w3c = (float*)(ws + OFF_W3);
  float* b3c = (float*)(ws + OFF_B3);
  const int tid = threadIdx.x;

  // W1 A-fragments: 4 m-tiles x 2 k-steps x 32 lanes x 16 values
  for (int e = tid; e < 4096; e += 256) {
    int m = e >> 10, rem = e & 1023;
    int ks = rem >> 9, lane = (rem >> 4) & 31, i = rem & 15;
    int half = lane >> 4, row = lane & 15, o = m * 16 + row;
    w1f[e] = f2bf(w1[o * 112 + afrag_k(ks, half, i)]);
  }
  // W2 A-fragments: 2 m-tiles x 2 k-steps
  for (int e = tid; e < 2048; e += 256) {
    int m = e >> 10, rem = e & 1023;
    int ks = rem >> 9, lane = (rem >> 4) & 31, i = rem & 15;
    int half = lane >> 4, row = lane & 15, o = m * 16 + row;
    w2f[e] = f2bf(w2[o * 64 + afrag_k(ks, half, i)]);
  }
  // PE contributions: enc(n)[j]=sin(n*div_j) (j<8) / cos (j>=8), div_j=10000^(-j/8)
  for (int e = tid; e < 128; e += 256) {           // d-term (+ b1 folded)
    int d = e >> 6, o = e & 63;
    float s = b1[o];
    #pragma unroll
    for (int j = 0; j < 8; ++j) {
      float dv = __powf(10000.f, -(float)j / 8.f);
      float t = (float)d * dv;
      s += w1[o * 112 + 64 + j] * __sinf(t) + w1[o * 112 + 72 + j] * __cosf(t);
    }
    ped[e] = s;
  }
  for (int e = tid; e < HH * 64; e += 256) {       // h-term
    int h = e >> 6, o = e & 63;
    float s = 0.f;
    #pragma unroll
    for (int j = 0; j < 8; ++j) {
      float dv = __powf(10000.f, -(float)j / 8.f);
      float t = (float)h * dv;
      s += w1[o * 112 + 80 + j] * __sinf(t) + w1[o * 112 + 88 + j] * __cosf(t);
    }
    peh[e] = s;
  }
  for (int e = tid; e < WW * 64; e += 256) {       // w-term
    int w = e >> 6, o = e & 63;
    float s = 0.f;
    #pragma unroll
    for (int j = 0; j < 8; ++j) {
      float dv = __powf(10000.f, -(float)j / 8.f);
      float t = (float)w * dv;
      s += w1[o * 112 + 96 + j] * __sinf(t) + w1[o * 112 + 104 + j] * __cosf(t);
    }
    pew[e] = s;
  }
  const float inv_sqrt = 1.f / sqrtf(1.f + 1e-5f);
  if (tid < 64) { s1[tid] = g1[tid] * inv_sqrt; t1[tid] = be1[tid]; }
  if (tid < 32) {
    float s = g2[tid] * inv_sqrt;
    s2[tid] = s; t2[tid] = be2[tid] + b2[tid] * s; w3c[tid] = w3[tid];
  }
  if (tid == 0) b3c[0] = b3[0];
}

// ---------------------------------------------------------------------------
// Kernel 1: dvf via bf16 WMMA.  grid = (275 tiles, N=4, src=2), 256 threads.
// src==0 -> writes d_out (f0), src==1 -> writes ws f1 buffer.
// ---------------------------------------------------------------------------
__global__ __launch_bounds__(256) void dvf_kernel(const float* __restrict__ x0,
                                                  const float* __restrict__ x1,
                                                  char* __restrict__ ws,
                                                  float* __restrict__ out) {
  __shared__ __align__(16) unsigned short Xs[ROWS * LDSTR];  // x bf16, then h1 bf16
  __shared__ float logitLds[ROWS];
  __shared__ float wtLds[ROWS];

  const int tid = threadIdx.x, lane = tid & 31, wave = tid >> 5;
  const int col = lane & 15, half = lane >> 4;
  const int n = blockIdx.y, src = blockIdx.z;
  const int tile0 = blockIdx.x * TILE;
  const float* __restrict__ x = src ? x1 : x0;

  const unsigned short* w1f = (const unsigned short*)(ws + OFF_W1F);
  const unsigned short* w2f = (const unsigned short*)(ws + OFF_W2F);
  const float* ped = (const float*)(ws + OFF_PED);
  const float* peh = (const float*)(ws + OFF_PEH);
  const float* pew = (const float*)(ws + OFF_PEW);
  const float* s1  = (const float*)(ws + OFF_S1);
  const float* t1  = (const float*)(ws + OFF_T1);
  const float* s2c = (const float*)(ws + OFF_S2);
  const float* t2c = (const float*)(ws + OFF_T2);
  const float* w3c = (const float*)(ws + OFF_W3);
  const float  b3v = *(const float*)(ws + OFF_B3);
  float* f1buf = (float*)(ws + OFF_F1);

  // --- Stage x tile into LDS as bf16, B-friendly [row][channel] layout -------
  for (int idx = tid; idx < 128 * TILE; idx += 256) {   // 128 (c,d) planes
    int plane = idx >> 7, p = idx & (TILE - 1);
    int c = plane >> 1, d = plane & 1;
    float v = x[(size_t)((n * CCH + c) * DD + d) * PP + tile0 + p];
    Xs[(p * 2 + d) * LDSTR + c] = f2bf(v);
  }
  logitLds[tid] = b3v;                                   // init logits with b3
  __syncthreads();

  // --- Layer 1: [256 rows] x [K=64] x [64 out] ; 64 tiles over 8 waves -------
  unsigned short h1u[64];
  #pragma unroll
  for (int t8 = 0; t8 < 8; ++t8) {
    const int t = t8 * 8 + wave, m = t >> 4, j = t & 15;
    v8f acc;
    #pragma unroll
    for (int q = 0; q < 8; ++q) acc[q] = 0.f;
    #pragma unroll
    for (int ks = 0; ks < 2; ++ks) {
      FragB16 A, B;
      const uint4* ap = (const uint4*)(w1f + ((size_t)(m * 2 + ks) * 32 + lane) * 16);
      A.q[0] = ap[0]; A.q[1] = ap[1];
      const uint4* bp = (const uint4*)(Xs + (j * 16 + col) * LDSTR + ks * 32 + half * 16);
      B.q[0] = bp[0]; B.q[1] = bp[1];
      acc = __builtin_amdgcn_wmma_f32_16x16x32_bf16(false, A.v, false, B.v,
                                                    (short)0, acc, false, false);
    }
    const int rrow = j * 16 + col;
    const int p = rrow >> 1, dd = rrow & 1;
    const int pos = tile0 + p, hh = pos / WW, wq = pos - hh * WW;
    #pragma unroll
    for (int q = 0; q < 8; ++q) {
      const int o = m * 16 + half * 8 + q;
      float pre = acc[q] + ped[dd * 64 + o] + peh[hh * 64 + o] + pew[wq * 64 + o];
      float y = pre * s1[o] + t1[o];
      y = (y >= 0.f) ? y : 0.01f * y;                    // LeakyReLU(0.01)
      h1u[t8 * 8 + q] = f2bf(y);
    }
  }
  __syncthreads();                                       // all B reads done -> reuse Xs
  #pragma unroll
  for (int t8 = 0; t8 < 8; ++t8) {
    const int t = t8 * 8 + wave, m = t >> 4, j = t & 15, rrow = j * 16 + col;
    #pragma unroll
    for (int q = 0; q < 8; ++q)
      Xs[rrow * LDSTR + (m * 16 + half * 8 + q)] = h1u[t8 * 8 + q];
  }
  __syncthreads();

  // --- Layer 2 + layer 3 reduction: 32 tiles over 8 waves --------------------
  #pragma unroll
  for (int t4 = 0; t4 < 4; ++t4) {
    const int t = t4 * 8 + wave, m = t >> 4, j = t & 15;
    v8f acc;
    #pragma unroll
    for (int q = 0; q < 8; ++q) acc[q] = 0.f;
    #pragma unroll
    for (int ks = 0; ks < 2; ++ks) {
      FragB16 A, B;
      const uint4* ap = (const uint4*)(w2f + ((size_t)(m * 2 + ks) * 32 + lane) * 16);
      A.q[0] = ap[0]; A.q[1] = ap[1];
      const uint4* bp = (const uint4*)(Xs + (j * 16 + col) * LDSTR + ks * 32 + half * 16);
      B.q[0] = bp[0]; B.q[1] = bp[1];
      acc = __builtin_amdgcn_wmma_f32_16x16x32_bf16(false, A.v, false, B.v,
                                                    (short)0, acc, false, false);
    }
    float part = 0.f;
    #pragma unroll
    for (int q = 0; q < 8; ++q) {
      const int o2 = m * 16 + half * 8 + q;
      float y = acc[q] * s2c[o2] + t2c[o2];
      y = (y >= 0.f) ? y : 0.01f * y;
      part += y * w3c[o2];                               // layer-3 dot (w3)
    }
    atomicAdd(&logitLds[j * 16 + col], part);            // ds_add_f32
  }
  __syncthreads();

  // --- Softmax over depth pair, then out = x * wts ---------------------------
  if (tid < TILE) {
    float l0 = logitLds[2 * tid], l1 = logitLds[2 * tid + 1];
    float mx = fmaxf(l0, l1);
    float e0 = __expf(l0 - mx), e1 = __expf(l1 - mx);
    float inv = 1.f / (e0 + e1);
    wtLds[2 * tid] = e0 * inv;
    wtLds[2 * tid + 1] = e1 * inv;
  }
  __syncthreads();

  float* __restrict__ dst = src ? f1buf : out;
  for (int idx = tid; idx < 128 * TILE; idx += 256) {
    int plane = idx >> 7, p = idx & (TILE - 1);
    int c = plane >> 1, d = plane & 1;
    float v = x[(size_t)((n * CCH + c) * DD + d) * PP + tile0 + p];
    dst[(size_t)(n * 128 + plane) * PP + tile0 + p] = v * wtLds[2 * p + d];
  }
}

// ---------------------------------------------------------------------------
// Kernel 2: rotate grid, bilinear-sample f1, max into d_out (which holds f0).
// ---------------------------------------------------------------------------
__global__ __launch_bounds__(256) void align_kernel(const float* __restrict__ tp,
                                                    const char* __restrict__ ws,
                                                    float* __restrict__ out) {
  const int pix = blockIdx.x * 256 + threadIdx.x;
  if (pix >= PP) return;
  const int n = blockIdx.y;
  const float* __restrict__ f1 = (const float*)(ws + OFF_F1);

  const int h = pix / WW, w = pix - h * WW;
  const float gx = 0.2f + 0.4f * (float)w;
  const float gy = -39.8f + 0.4f * (float)h;
  const float thp = tp[n * 2 + 0], thc = tp[n * 2 + 1];
  float c1 = __cosf(thc), s1v = __sinf(thc);
  float xr = gx * c1 - gy * s1v;
  float yr = gx * s1v + gy * c1;
  float c2 = __cosf(thp), s2v = __sinf(thp);       // rot by -th_prev
  float x2 = xr * c2 + yr * s2v;
  float y2 = -xr * s2v + yr * c2;
  float xi = x2 * 2.5f;                            // (x - 0) / 0.05 / 8
  float yi = (y2 + 40.f) * 2.5f;

  int xf = (int)floorf(xi), yf = (int)floorf(yi);
  int x0i = min(max(xf, 0), WW - 1), x1i = min(max(xf + 1, 0), WW - 1);
  int y0i = min(max(yf, 0), HH - 1), y1i = min(max(yf + 1, 0), HH - 1);
  float wa = ((float)x1i - xi) * ((float)y1i - yi);
  float wb = ((float)x1i - xi) * (yi - (float)y0i);
  float wc = (xi - (float)x0i) * ((float)y1i - yi);
  float wd = (xi - (float)x0i) * (yi - (float)y0i);

  const int ia = y0i * WW + x0i, ib = y1i * WW + x0i;
  const int ic = y0i * WW + x1i, id = y1i * WW + x1i;
  for (int ch = 0; ch < 128; ++ch) {
    const size_t base = (size_t)(n * 128 + ch) * PP;
    float s = f1[base + ia] * wa + f1[base + ib] * wb +
              f1[base + ic] * wc + f1[base + id] * wd;
    const size_t oi = base + pix;
    out[oi] = fmaxf(out[oi], s);
  }
}

// ---------------------------------------------------------------------------
extern "C" void kernel_launch(void* const* d_in, const int* in_sizes, int n_in,
                              void* d_out, int out_size, void* d_ws, size_t ws_size,
                              hipStream_t stream) {
  const float* x0  = (const float*)d_in[0];
  const float* x1  = (const float*)d_in[1];
  const float* tp  = (const float*)d_in[2];
  const float* w1  = (const float*)d_in[3];
  const float* b1  = (const float*)d_in[4];
  const float* g1  = (const float*)d_in[5];
  const float* be1 = (const float*)d_in[6];
  const float* w2  = (const float*)d_in[7];
  const float* b2  = (const float*)d_in[8];
  const float* g2  = (const float*)d_in[9];
  const float* be2 = (const float*)d_in[10];
  const float* w3  = (const float*)d_in[11];
  const float* b3  = (const float*)d_in[12];
  float* out = (float*)d_out;
  char* ws = (char*)d_ws;

  prep_kernel<<<1, 256, 0, stream>>>(w1, b1, g1, be1, w2, b2, g2, be2, w3, b3, ws);
  dvf_kernel<<<dim3(PP / TILE, NB, 2), 256, 0, stream>>>(x0, x1, ws, out);
  align_kernel<<<dim3((PP + 255) / 256, NB), 256, 0, stream>>>(tp, ws, out);
}
